// TripleAttention_30571577213071
// MI455X (gfx1250) — compile-verified, matrix operand order checked by reading, not run on hardware
//
#include <hip/hip_runtime.h>
#include <hip/hip_bf16.h>
#include <math.h>

// ---------------------------------------------------------------------------
// Problem constants (from reference): N=32768, M=4096, C=128, H=8, K=16,
// HID=512, D=16.
// ---------------------------------------------------------------------------
#define NN 32768
#define MM 4096
#define CC 128
#define KK 16
#define HID 512

typedef __attribute__((ext_vector_type(16))) __bf16 v16bf;
typedef __attribute__((ext_vector_type(8)))  float  v8f;

// ---------------------------------------------------------------------------
// WMMA helpers (CDNA5 wave32, V_WMMA_F32_16X16X32_BF16).
// A (16x32 bf16): lane&15 = row M; lanes<16 hold K {0..7,16..23},
// lanes>=16 hold K {8..15,24..31}, 2 K's per VGPR (ISA 7.12.2).
// B (32x16 bf16): symmetric with N on lane&15.
// C/D (16x16 f32): VGPR i -> row i + 8*(lane>>4), col = lane&15.
// ---------------------------------------------------------------------------
__device__ __forceinline__ int kmap(int v, int hi) {
  return (v < 4 ? 2 * v : 16 + 2 * (v - 4)) + (hi ? 8 : 0);
}

__device__ __forceinline__ v8f wmma_bf16(v16bf a, v16bf b, v8f c) {
  return __builtin_amdgcn_wmma_f32_16x16x32_bf16(
      /*neg_a=*/false, a, /*neg_b=*/false, b,
      /*c_mod=*/(short)0, c, /*reuse_a=*/false, /*reuse_b=*/false);
}

// A fragment from a row-major f32 tile (16 rows, leading dim lda), k window k0.
__device__ __forceinline__ v16bf frag_a(const float* A, int lda, int k0) {
  int lane = threadIdx.x & 31;
  int r = lane & 15, hi = lane >> 4;
  v16bf f;
#pragma unroll
  for (int v = 0; v < 8; ++v) {
    int k = kmap(v, hi);
    f[2 * v]     = (__bf16)A[r * lda + k0 + k];
    f[2 * v + 1] = (__bf16)A[r * lda + k0 + k + 1];
  }
  return f;
}

// B fragment from a row-major f32 weight W[k][n] with leading dim ldw,
// k window k0, column base n0.
__device__ __forceinline__ v16bf frag_b(const float* W, int ldw, int k0, int n0) {
  int lane = threadIdx.x & 31;
  int c = lane & 15, hi = lane >> 4;
  v16bf f;
#pragma unroll
  for (int v = 0; v < 8; ++v) {
    int k = kmap(v, hi);
    f[2 * v]     = (__bf16)W[(size_t)(k0 + k) * ldw + n0 + c];
    f[2 * v + 1] = (__bf16)W[(size_t)(k0 + k + 1) * ldw + n0 + c];
  }
  return f;
}

__device__ __forceinline__ float gelu_f(float v) {
  return 0.5f * v * (1.0f + erff(v * 0.70710678118654752f));
}

// ---------------------------------------------------------------------------
// Async global -> LDS tile staging (CDNA5 GLOBAL_LOAD_ASYNC_TO_LDS_B128,
// GVS mode: mem = SADDR + VADDR + IOFFSET; VDST VGPR = LDS byte address).
// Tracked by ASYNCcnt; drained with s_wait_asynccnt before the barrier.
// BYTES must be a multiple of 16 for full coverage.
// ---------------------------------------------------------------------------
__device__ __forceinline__ void async_stage(const float* __restrict__ src,
                                            float* dstLds, int bytes) {
  unsigned lds0 = (unsigned)(uintptr_t)dstLds;  // low 32 bits = LDS byte addr
  for (int t = (int)threadIdx.x * 16; t < bytes; t += (int)blockDim.x * 16) {
    unsigned ldsAddr = lds0 + (unsigned)t;
    int voff = t;
    asm volatile("global_load_async_to_lds_b128 %0, %1, %2"
                 :
                 : "v"(ldsAddr), "v"(voff), "s"(src)
                 : "memory");
  }
  asm volatile("s_wait_asynccnt 0x0" ::: "memory");
}

// ---------------------------------------------------------------------------
// LayerNorm over 128 channels, one wave per row (wave32: 4 ch per lane).
// ---------------------------------------------------------------------------
__device__ __forceinline__ void ln_row(const float4 v, float4* o,
                                       const float* g, const float* b, int lane) {
  float s = v.x + v.y + v.z + v.w;
#pragma unroll
  for (int m = 16; m >= 1; m >>= 1) s += __shfl_xor(s, m, 32);
  float mu = s * (1.0f / 128.0f);
  float dx = v.x - mu, dy = v.y - mu, dz = v.z - mu, dw = v.w - mu;
  float q = dx * dx + dy * dy + dz * dz + dw * dw;
#pragma unroll
  for (int m = 16; m >= 1; m >>= 1) q += __shfl_xor(q, m, 32);
  float inv = rsqrtf(q * (1.0f / 128.0f) + 1e-5f);
  float4 gg = *(const float4*)(g + lane * 4);
  float4 bb = *(const float4*)(b + lane * 4);
  o->x = dx * inv * gg.x + bb.x;
  o->y = dy * inv * gg.y + bb.y;
  o->z = dz * inv * gg.z + bb.z;
  o->w = dw * inv * gg.w + bb.w;
}

__global__ void ln_kernel(const float* __restrict__ in, float* __restrict__ out,
                          const float* __restrict__ g, const float* __restrict__ b,
                          int rows) {
  int wid = (int)((blockIdx.x * blockDim.x + threadIdx.x) >> 5);
  if (wid >= rows) return;
  int lane = threadIdx.x & 31;
  float4 v = *(const float4*)(in + (size_t)wid * CC + lane * 4);
  float4 o;
  ln_row(v, &o, g, b, lane);
  *(float4*)(out + (size_t)wid * CC + lane * 4) = o;
}

// gx = x[knn_idx] (gather) + LN(gx) in one pass.
__global__ void gather_ln_kernel(const float* __restrict__ x,
                                 const int* __restrict__ idx,
                                 float* __restrict__ gx, float* __restrict__ lnout,
                                 const float* __restrict__ g,
                                 const float* __restrict__ b, int rows) {
  int wid = (int)((blockIdx.x * blockDim.x + threadIdx.x) >> 5);
  if (wid >= rows) return;
  int lane = threadIdx.x & 31;
  int j = idx[wid];
  float4 v = *(const float4*)(x + (size_t)j * CC + lane * 4);
  *(float4*)(gx + (size_t)wid * CC + lane * 4) = v;
  float4 o;
  ln_row(v, &o, g, b, lane);
  *(float4*)(lnout + (size_t)wid * CC + lane * 4) = o;
}

// ---------------------------------------------------------------------------
// Generic WMMA linear: out = act(in @ W + bias) [+ residual]
// 16 rows per block (8 waves), A tile async-staged into LDS.
// Compile-time CIN/COUT -> fully unrolled k-loops and constant LDS offsets.
// ---------------------------------------------------------------------------
template <int CIN, int COUT, int ACT>
__global__ void linear_kernel(const float* __restrict__ in, const float* __restrict__ W,
                              const float* __restrict__ bias,
                              const float* __restrict__ residual,
                              float* __restrict__ out) {
  __shared__ float Atile[16 * CIN];
  int tile = blockIdx.x;
  __builtin_prefetch(W, 0, 1);  // global_prefetch_b8: pull weights toward GL2
  async_stage(in + (size_t)tile * 16 * CIN, Atile, 16 * CIN * 4);
  __syncthreads();
  int w = threadIdx.x >> 5, lane = threadIdx.x & 31;
  int col = lane & 15, hi = lane >> 4;
#pragma unroll
  for (int ct = 0; ct < COUT / 16 / 8; ++ct) {
    int n0 = (ct * 8 + w) * 16;
    v8f acc = {};
#pragma unroll
    for (int k0 = 0; k0 < CIN; k0 += 32)
      acc = wmma_bf16(frag_a(Atile, CIN, k0), frag_b(W, COUT, k0, n0), acc);
#pragma unroll
    for (int i = 0; i < 8; ++i) {
      int row = i + hi * 8;
      int n = n0 + col;
      float vv = acc[i] + bias[n];
      if (ACT == 1) vv = gelu_f(vv);
      size_t o = ((size_t)tile * 16 + row) * (size_t)COUT + n;
      if (residual) vv += residual[o];
      out[o] = vv;
    }
  }
  // remainder column tiles when COUT/16 not a multiple of 8
#pragma unroll
  for (int ct = (COUT / 16 / 8) * 8 + w; ct < COUT / 16; ct += 8) {
    int n0 = ct * 16;
    v8f acc = {};
#pragma unroll
    for (int k0 = 0; k0 < CIN; k0 += 32)
      acc = wmma_bf16(frag_a(Atile, CIN, k0), frag_b(W, COUT, k0, n0), acc);
#pragma unroll
    for (int i = 0; i < 8; ++i) {
      int row = i + hi * 8;
      int n = n0 + col;
      float vv = acc[i] + bias[n];
      if (ACT == 1) vv = gelu_f(vv);
      size_t o = ((size_t)tile * 16 + row) * (size_t)COUT + n;
      if (residual) vv += residual[o];
      out[o] = vv;
    }
  }
}

// ---------------------------------------------------------------------------
// Fused MLP: x += fc2(gelu(fc1(ln_x))); 128 -> 512 -> 128, hidden in LDS.
// ---------------------------------------------------------------------------
__global__ void mlp_kernel(float* __restrict__ x, const float* __restrict__ lnx,
                           const float* __restrict__ W1, const float* __restrict__ b1,
                           const float* __restrict__ W2, const float* __restrict__ b2) {
  __shared__ float A[16 * CC];    // 8 KB
  __shared__ float Hh[16 * HID];  // 32 KB
  int tile = blockIdx.x;
  __builtin_prefetch(W1, 0, 1);
  __builtin_prefetch(W2, 0, 1);
  async_stage(lnx + (size_t)tile * 16 * CC, A, 16 * CC * 4);
  __syncthreads();
  int w = threadIdx.x >> 5, lane = threadIdx.x & 31;
  int col = lane & 15, hi = lane >> 4;
#pragma unroll
  for (int c4 = 0; c4 < 4; ++c4) {  // 4 column tiles per wave (HID/16/8)
    int ct = c4 * 8 + w;
    v8f acc = {};
#pragma unroll
    for (int k0 = 0; k0 < CC; k0 += 32)
      acc = wmma_bf16(frag_a(A, CC, k0), frag_b(W1, HID, k0, ct * 16), acc);
#pragma unroll
    for (int i = 0; i < 8; ++i) {
      int row = i + hi * 8, n = ct * 16 + col;
      Hh[row * HID + n] = gelu_f(acc[i] + b1[n]);
    }
  }
  __syncthreads();
  {
    int ct = w;  // one 16-col tile per wave (8*16 = 128)
    v8f acc = {};
#pragma unroll
    for (int k0 = 0; k0 < HID; k0 += 32)
      acc = wmma_bf16(frag_a(Hh, HID, k0), frag_b(W2, CC, k0, ct * 16), acc);
#pragma unroll
    for (int i = 0; i < 8; ++i) {
      int row = i + hi * 8, n = ct * 16 + col;
      size_t o = ((size_t)tile * 16 + row) * CC + n;
      x[o] = x[o] + acc[i] + b2[n];
    }
  }
}

// ---------------------------------------------------------------------------
// Local grouped attention (_lga), fully fused per group of K=16 rows.
// qkv GEMM + pe + per-head (wave-per-head) WMMA attention + proj + residual.
// ---------------------------------------------------------------------------
__global__ void lga_kernel(float* __restrict__ gx, const float* __restrict__ lnx,
                           const float* __restrict__ p, const float* __restrict__ n_p,
                           const int* __restrict__ knn_idx,
                           const float* __restrict__ Wqkv, const float* __restrict__ bqkv,
                           const float* __restrict__ Wpe, const float* __restrict__ bpe,
                           const float* __restrict__ Wproj, const float* __restrict__ bproj) {
  __shared__ float A[16 * CC];        // ln(gx) tile
  __shared__ float QKV[16 * 384];
  __shared__ float PE[16 * CC];
  __shared__ float ATT[16 * CC];
  __shared__ float PR[16][3];
  __shared__ float PROBS[8][16 * 16];
  int m = blockIdx.x;
  int tid = threadIdx.x;
  __builtin_prefetch(Wqkv, 0, 1);
  async_stage(lnx + (size_t)m * 16 * CC, A, 16 * CC * 4);
  if (tid < 16) {
    int j = knn_idx[m * KK + tid];
    PR[tid][0] = p[j * 3 + 0] - n_p[m * 3 + 0];
    PR[tid][1] = p[j * 3 + 1] - n_p[m * 3 + 1];
    PR[tid][2] = p[j * 3 + 2] - n_p[m * 3 + 2];
  }
  __syncthreads();
  int w = tid >> 5, lane = tid & 31, col = lane & 15, hi = lane >> 4;

  // qkv: 16x128 @ 128x384 (3 col tiles per wave)
#pragma unroll
  for (int c3 = 0; c3 < 3; ++c3) {
    int ct = c3 * 8 + w;
    v8f acc = {};
#pragma unroll
    for (int k0 = 0; k0 < CC; k0 += 32)
      acc = wmma_bf16(frag_a(A, CC, k0), frag_b(Wqkv, 384, k0, ct * 16), acc);
#pragma unroll
    for (int i = 0; i < 8; ++i) {
      int row = i + hi * 8, n = ct * 16 + col;
      QKV[row * 384 + n] = acc[i] + bqkv[n];
    }
  }
  // pe: 16x3 @ 3x128 (tiny -> scalar VALU)
  for (int t = tid; t < 16 * CC; t += 256) {
    int r = t >> 7, c = t & 127;
    PE[t] = bpe[c] + PR[r][0] * Wpe[c] + PR[r][1] * Wpe[CC + c] + PR[r][2] * Wpe[2 * CC + c];
  }
  __syncthreads();

  // scores = q @ (k+pe)^T * D^-0.5  (wave w == head w; K padded 16->32)
  {
    int h = w;
    v16bf af, bf;
    int r = lane & 15;
#pragma unroll
    for (int v = 0; v < 8; ++v) {
      int k = kmap(v, hi);
      af[2 * v]     = (k < 16)     ? (__bf16)QKV[r * 384 + h * 16 + k]     : (__bf16)0.0f;
      af[2 * v + 1] = (k + 1 < 16) ? (__bf16)QKV[r * 384 + h * 16 + k + 1] : (__bf16)0.0f;
      // B[kdim][n] = K[n][kdim] + PE[n][kdim]  (transposed key), n = r
      bf[2 * v]     = (k < 16)
          ? (__bf16)(QKV[r * 384 + 128 + h * 16 + k] + PE[r * CC + h * 16 + k]) : (__bf16)0.0f;
      bf[2 * v + 1] = (k + 1 < 16)
          ? (__bf16)(QKV[r * 384 + 128 + h * 16 + k + 1] + PE[r * CC + h * 16 + k + 1]) : (__bf16)0.0f;
    }
    v8f sc = {};
    sc = wmma_bf16(af, bf, sc);
    // softmax across the 16 columns (lanes 0..15 of each half-wave)
    float* PB = &PROBS[w][0];
#pragma unroll
    for (int i = 0; i < 8; ++i) {
      float s = sc[i] * 0.25f;  // D^-0.5 = 1/4
      float mx = s;
#pragma unroll
      for (int mm = 8; mm >= 1; mm >>= 1) mx = fmaxf(mx, __shfl_xor(mx, mm, 32));
      float e = __expf(s - mx);
      float sum = e;
#pragma unroll
      for (int mm = 8; mm >= 1; mm >>= 1) sum += __shfl_xor(sum, mm, 32);
      PB[(i + hi * 8) * 16 + col] = e / sum;
    }
  }
  __syncthreads();

  // out_h = probs @ (v+pe)
  {
    int h = w;
    const float* PB = &PROBS[w][0];
    v16bf af, bf;
    int r = lane & 15;
#pragma unroll
    for (int v = 0; v < 8; ++v) {
      int k = kmap(v, hi);
      af[2 * v]     = (k < 16)     ? (__bf16)PB[r * 16 + k]     : (__bf16)0.0f;
      af[2 * v + 1] = (k + 1 < 16) ? (__bf16)PB[r * 16 + k + 1] : (__bf16)0.0f;
      // B[j][n] = V[j][n] + PE[j][n], row j = K index, n = output dim = r
      bf[2 * v]     = (k < 16)
          ? (__bf16)(QKV[k * 384 + 256 + h * 16 + r] + PE[k * CC + h * 16 + r]) : (__bf16)0.0f;
      bf[2 * v + 1] = (k + 1 < 16)
          ? (__bf16)(QKV[(k + 1) * 384 + 256 + h * 16 + r] + PE[(k + 1) * CC + h * 16 + r]) : (__bf16)0.0f;
    }
    v8f o = {};
    o = wmma_bf16(af, bf, o);
#pragma unroll
    for (int i = 0; i < 8; ++i) ATT[(i + hi * 8) * CC + h * 16 + col] = o[i];
  }
  __syncthreads();

  // proj (128->128) + residual, in place into gx
  {
    int ct = w;
    v8f acc = {};
#pragma unroll
    for (int k0 = 0; k0 < CC; k0 += 32)
      acc = wmma_bf16(frag_a(ATT, CC, k0), frag_b(Wproj, CC, k0, ct * 16), acc);
#pragma unroll
    for (int i = 0; i < 8; ++i) {
      int row = i + hi * 8, n = ct * 16 + col;
      size_t o = ((size_t)m * 16 + row) * CC + n;
      gx[o] = gx[o] + acc[i] + bproj[n];
    }
  }
}

// ---------------------------------------------------------------------------
// Max-pool over the K=16 neighbors of each group.
// ---------------------------------------------------------------------------
__global__ void maxpool_kernel(const float* __restrict__ gx, float* __restrict__ xm) {
  int m = blockIdx.x, c = threadIdx.x;
  float v = -3.402823466e38f;
  for (int k = 0; k < KK; ++k) v = fmaxf(v, gx[((size_t)m * KK + k) * CC + c]);
  xm[(size_t)m * CC + c] = v;
}

// ---------------------------------------------------------------------------
// Brute-force KNN: one thread per query, insertion top-16 (smallest d2,
// earliest index wins ties, matching top_k(-d2)). Point set is 48 KB -> L2.
// ---------------------------------------------------------------------------
__global__ void knn_kernel(const float* __restrict__ qp, int nq,
                           const float* __restrict__ kp, int np,
                           int* __restrict__ idx) {
  int i = blockIdx.x * blockDim.x + threadIdx.x;
  if (i >= nq) return;
  float qx = qp[i * 3], qy = qp[i * 3 + 1], qz = qp[i * 3 + 2];
  float bd[16];
  int bi[16];
#pragma unroll
  for (int t = 0; t < 16; ++t) { bd[t] = 3.402823466e38f; bi[t] = 0; }
  for (int j = 0; j < np; ++j) {
    float dx = qx - kp[j * 3], dy = qy - kp[j * 3 + 1], dz = qz - kp[j * 3 + 2];
    float d = dx * dx + dy * dy + dz * dz;
    if (d < bd[15]) {
      int t = 15;
      while (t > 0 && bd[t - 1] > d) { bd[t] = bd[t - 1]; bi[t] = bi[t - 1]; --t; }
      bd[t] = d; bi[t] = j;
    }
  }
#pragma unroll
  for (int t = 0; t < 16; ++t) idx[(size_t)i * 16 + t] = bi[t];
}

// ---------------------------------------------------------------------------
// Scalar KNN-attention (_attend with gathered k/v + pe). One thread per
// (query, head): K=16 neighbors, D=16. Gather/latency bound, not matrix bound.
// ---------------------------------------------------------------------------
__global__ void attend_kernel(const float* __restrict__ qsrc, int qld, int qoff,
                              const float* __restrict__ ksrc, int kld, int koff,
                              const float* __restrict__ vsrc, int vld, int voff,
                              const int* __restrict__ idx,
                              const float* __restrict__ posq,
                              const float* __restrict__ posp,
                              const float* __restrict__ Wpe, const float* __restrict__ bpe,
                              float* __restrict__ out, int nq) {
  int t = blockIdx.x * blockDim.x + threadIdx.x;
  if (t >= nq * 8) return;
  int i = t >> 3, h = t & 7;
  float q[16];
#pragma unroll
  for (int d = 0; d < 16; ++d) q[d] = qsrc[(size_t)i * qld + qoff + h * 16 + d];
  float px = posq[i * 3], py = posq[i * 3 + 1], pz = posq[i * 3 + 2];
  float s[16];
  int nb[16];
  float mx = -3.402823466e38f;
  for (int j = 0; j < 16; ++j) {
    int jj = idx[(size_t)i * 16 + j];
    nb[j] = jj;
    float dx = posp[jj * 3] - px, dy = posp[jj * 3 + 1] - py, dz = posp[jj * 3 + 2] - pz;
    float acc = 0.f;
#pragma unroll
    for (int d = 0; d < 16; ++d) {
      int c = h * 16 + d;
      float pe = bpe[c] + dx * Wpe[c] + dy * Wpe[CC + c] + dz * Wpe[2 * CC + c];
      acc += q[d] * (ksrc[(size_t)jj * kld + koff + c] + pe);
    }
    s[j] = acc * 0.25f;
    mx = fmaxf(mx, s[j]);
  }
  float sum = 0.f;
#pragma unroll
  for (int j = 0; j < 16; ++j) { s[j] = __expf(s[j] - mx); sum += s[j]; }
  float inv = 1.f / sum;
  float o[16];
#pragma unroll
  for (int d = 0; d < 16; ++d) o[d] = 0.f;
  for (int j = 0; j < 16; ++j) {
    int jj = nb[j];
    float a = s[j] * inv;
    float dx = posp[jj * 3] - px, dy = posp[jj * 3 + 1] - py, dz = posp[jj * 3 + 2] - pz;
#pragma unroll
    for (int d = 0; d < 16; ++d) {
      int c = h * 16 + d;
      float pe = bpe[c] + dx * Wpe[c] + dy * Wpe[CC + c] + dz * Wpe[2 * CC + c];
      o[d] += a * (vsrc[(size_t)jj * vld + voff + c] + pe);
    }
  }
#pragma unroll
  for (int d = 0; d < 16; ++d) out[(size_t)i * CC + h * 16 + d] = o[d];
}

// ---------------------------------------------------------------------------
// Host side. Input flattening assumption: top-level setup_inputs() dict in
// insertion order (p, x, n_p, knn_idx, params), with `params` flattened by
// jax tree_leaves (keys sorted recursively; lin -> {b, w}; norm -> {b, g}).
// ---------------------------------------------------------------------------
extern "C" void kernel_launch(void* const* d_in, const int* in_sizes, int n_in,
                              void* d_out, int out_size, void* d_ws, size_t ws_size,
                              hipStream_t stream) {
  const float* p     = (const float*)d_in[0];   // (N,3)
  const float* x     = (const float*)d_in[1];   // (N,C)
  const float* n_p   = (const float*)d_in[2];   // (M,3)
  const int*   knn   = (const int*)d_in[3];     // (M,K)
  // params, sorted leaves starting at index 4:
  const float* ca_kv_b   = (const float*)d_in[4];
  const float* ca_kv_w   = (const float*)d_in[5];
  const float* ca_pe_b   = (const float*)d_in[6];
  const float* ca_pe_w   = (const float*)d_in[7];
  const float* ca_pj_b   = (const float*)d_in[8];
  const float* ca_pj_w   = (const float*)d_in[9];
  const float* ca_q_b    = (const float*)d_in[10];
  const float* ca_q_w    = (const float*)d_in[11];
  const float* cm_f1_b   = (const float*)d_in[12];
  const float* cm_f1_w   = (const float*)d_in[13];
  const float* cm_f2_b   = (const float*)d_in[14];
  const float* cm_f2_w   = (const float*)d_in[15];
  const float* cn1_b     = (const float*)d_in[16];
  const float* cn1_g     = (const float*)d_in[17];
  const float* cn2_b     = (const float*)d_in[18];
  const float* cn2_g     = (const float*)d_in[19];
  const float* cn3_b     = (const float*)d_in[20];
  const float* cn3_g     = (const float*)d_in[21];
  const float* ga_pe_b   = (const float*)d_in[22];
  const float* ga_pe_w   = (const float*)d_in[23];
  const float* ga_pj_b   = (const float*)d_in[24];
  const float* ga_pj_w   = (const float*)d_in[25];
  const float* ga_qkv_b  = (const float*)d_in[26];
  const float* ga_qkv_w  = (const float*)d_in[27];
  const float* gm_f1_b   = (const float*)d_in[28];
  const float* gm_f1_w   = (const float*)d_in[29];
  const float* gm_f2_b   = (const float*)d_in[30];
  const float* gm_f2_w   = (const float*)d_in[31];
  const float* gn1_b     = (const float*)d_in[32];
  const float* gn1_g     = (const float*)d_in[33];
  const float* gn2_b     = (const float*)d_in[34];
  const float* gn2_g     = (const float*)d_in[35];
  const float* la_pe_b   = (const float*)d_in[36];
  const float* la_pe_w   = (const float*)d_in[37];
  const float* la_pj_b   = (const float*)d_in[38];
  const float* la_pj_w   = (const float*)d_in[39];
  const float* la_qkv_b  = (const float*)d_in[40];
  const float* la_qkv_w  = (const float*)d_in[41];
  const float* lm_f1_b   = (const float*)d_in[42];
  const float* lm_f1_w   = (const float*)d_in[43];
  const float* lm_f2_b   = (const float*)d_in[44];
  const float* lm_f2_w   = (const float*)d_in[45];
  const float* ln1_b     = (const float*)d_in[46];
  const float* ln1_g     = (const float*)d_in[47];
  const float* ln2_b     = (const float*)d_in[48];
  const float* ln2_g     = (const float*)d_in[49];

  float* out = (float*)d_out;
  const int MK = MM * KK;  // 65536 grouped rows

  // Workspace carve-out (~137 MB total).
  char* ws = (char*)d_ws;
  size_t off = 0;
  auto carve = [&](size_t bytes) -> void* {
    void* r = ws + off;
    off = (off + bytes + 255) & ~(size_t)255;
    return r;
  };
  float* gx     = (float*)carve((size_t)MK * CC * 4);   // 32 MB
  float* lnbuf  = (float*)carve((size_t)MK * CC * 4);   // 32 MB (reused scratch)
  float* qkv_g  = (float*)carve((size_t)MM * 384 * 4);  // 6 MB
  float* attn_g = (float*)carve((size_t)MM * CC * 4);   // 2 MB
  float* xm     = (float*)carve((size_t)MM * CC * 4);   // 2 MB
  float* xm2    = (float*)carve((size_t)MM * CC * 4);   // 2 MB
  float* feat   = (float*)carve((size_t)NN * CC * 4);   // 16 MB
  float* qbuf   = (float*)carve((size_t)NN * CC * 4);   // 16 MB
  float* kvbuf  = (float*)carve((size_t)MM * 256 * 4);  // 4 MB
  float* attn_c = (float*)carve((size_t)NN * CC * 4);   // 16 MB
  int*   idx_g  = (int*)carve((size_t)MM * KK * 4);
  int*   idx_c  = (int*)carve((size_t)NN * KK * 4);
  (void)in_sizes; (void)n_in; (void)out_size; (void)ws_size;

  // ---- Local stage ----
  gather_ln_kernel<<<MK * 32 / 256, 256, 0, stream>>>(x, knn, gx, lnbuf, ln1_g, ln1_b, MK);
  lga_kernel<<<MM, 256, 0, stream>>>(gx, lnbuf, p, n_p, knn,
                                     la_qkv_w, la_qkv_b, la_pe_w, la_pe_b, la_pj_w, la_pj_b);
  ln_kernel<<<MK * 32 / 256, 256, 0, stream>>>(gx, lnbuf, ln2_g, ln2_b, MK);
  mlp_kernel<<<MK / 16, 256, 0, stream>>>(gx, lnbuf, lm_f1_w, lm_f1_b, lm_f2_w, lm_f2_b);

  // ---- Global stage ----
  maxpool_kernel<<<MM, CC, 0, stream>>>(gx, xm);
  knn_kernel<<<(MM + 255) / 256, 256, 0, stream>>>(n_p, MM, n_p, MM, idx_g);
  ln_kernel<<<MM * 32 / 256, 256, 0, stream>>>(xm, lnbuf, gn1_g, gn1_b, MM);
  linear_kernel<CC, 384, 0><<<MM / 16, 256, 0, stream>>>(lnbuf, ga_qkv_w, ga_qkv_b,
                                                         nullptr, qkv_g);
  attend_kernel<<<MM * 8 / 256, 256, 0, stream>>>(qkv_g, 384, 0, qkv_g, 384, 128,
                                                  qkv_g, 384, 256, idx_g, n_p, n_p,
                                                  ga_pe_w, ga_pe_b, attn_g, MM);
  linear_kernel<CC, CC, 0><<<MM / 16, 256, 0, stream>>>(attn_g, ga_pj_w, ga_pj_b, xm, xm);
  ln_kernel<<<MM * 32 / 256, 256, 0, stream>>>(xm, lnbuf, gn2_g, gn2_b, MM);
  mlp_kernel<<<MM / 16, 256, 0, stream>>>(xm, lnbuf, gm_f1_w, gm_f1_b, gm_f2_w, gm_f2_b);

  // ---- Cross stage ----
  ln_kernel<<<NN * 32 / 256, 256, 0, stream>>>(x, feat, cn1_g, cn1_b, NN);
  ln_kernel<<<MM * 32 / 256, 256, 0, stream>>>(xm, xm2, cn2_g, cn2_b, MM);
  knn_kernel<<<(NN + 255) / 256, 256, 0, stream>>>(p, NN, n_p, MM, idx_c);
  linear_kernel<CC, CC, 0><<<NN / 16, 256, 0, stream>>>(feat, ca_q_w, ca_q_b,
                                                        nullptr, qbuf);
  linear_kernel<CC, 256, 0><<<MM / 16, 256, 0, stream>>>(xm2, ca_kv_w, ca_kv_b,
                                                         nullptr, kvbuf);
  attend_kernel<<<NN * 8 / 256, 256, 0, stream>>>(qbuf, 128, 0, kvbuf, 256, 0,
                                                  kvbuf, 256, 128, idx_c, p, n_p,
                                                  ca_pe_w, ca_pe_b, attn_c, NN);
  linear_kernel<CC, CC, 0><<<NN / 16, 256, 0, stream>>>(attn_c, ca_pj_w, ca_pj_b, x, out);
  ln_kernel<<<NN * 32 / 256, 256, 0, stream>>>(out, lnbuf, cn3_g, cn3_b, NN);
  mlp_kernel<<<NN / 16, 256, 0, stream>>>(out, lnbuf, cm_f1_w, cm_f1_b, cm_f2_w, cm_f2_b);
}
